// RelativeMultiHeadAttentionLayer_7937099563620
// MI455X (gfx1250) — compile-verified, matrix operand order checked by reading, not run on hardware
//
#include <hip/hip_runtime.h>

typedef __attribute__((ext_vector_type(16))) __bf16 v16bf;
typedef __attribute__((ext_vector_type(8)))  float  v8f;
typedef __attribute__((ext_vector_type(4)))  unsigned int u32x4;

#define T_  1024
#define C_  192
#define B_  4
#define H_  8
#define D_  24
#define DP  32      // padded head dim for WMMA (K=32)
#define KST 40      // bf16 per K row in LDS (bank-conflict pad, 16B aligned)
#define VST 1040    // bf16 per vt row in LDS

// Two aligned 16B chunks -> one 16 x bf16 WMMA operand (8 VGPRs)
__device__ __forceinline__ v16bf ld2x16(const __bf16* lo, const __bf16* hi) {
  union { u32x4 q[2]; v16bf v; } u;
  u.q[0] = *(const u32x4*)lo;
  u.q[1] = *(const u32x4*)hi;
  return u.v;
}

__global__ void zero_u32(unsigned* __restrict__ p, int n) {
  int i = blockIdx.x * blockDim.x + threadIdx.x;
  if (i < n) p[i] = 0u;
}

// f32 -> bf16 for the four 192x192 weight matrices (row-major preserved)
__global__ void cvt_w(const float* __restrict__ w0, const float* __restrict__ w1,
                      const float* __restrict__ w2, const float* __restrict__ w3,
                      __bf16* __restrict__ out) {
  int i = blockIdx.x * blockDim.x + threadIdx.x;
  if (i >= 4 * C_ * C_) return;
  const int m = i / (C_ * C_), r = i % (C_ * C_);
  const float* w = (m == 0) ? w0 : (m == 1) ? w1 : (m == 2) ? w2 : w3;
  out[i] = (__bf16)w[r];
}

// x[b][c][t] f32 -> xt[b][t][192] bf16 (transpose + convert, done once)
__global__ void cvt_x(const float* __restrict__ x, __bf16* __restrict__ xt) {
  int i = blockIdx.x * blockDim.x + threadIdx.x;
  if (i >= B_ * C_ * T_) return;
  const int b = i / (C_ * T_);
  const int rem = i % (C_ * T_);
  const int c = rem / T_, t = rem % T_;
  xt[(b * T_ + t) * C_ + c] = (__bf16)x[i];
}

// mask[b][t][s] int32 -> bitmask mkb[b][s][t/32], bit j of word = row t=tw*32+j.
// Input reads coalesced across s (adjacent threads), 16 MB read once.
__global__ void pack_mask(const int* __restrict__ mask, unsigned* __restrict__ mkb) {
  int i = blockIdx.x * blockDim.x + threadIdx.x;    // (b, tw, s), s fastest
  if (i >= B_ * T_ * 32) return;
  const int s  = i & (T_ - 1);
  const int tw = (i >> 10) & 31;
  const int b  = i >> 15;
  const int* mrow = mask + ((size_t)b * T_ + tw * 32) * T_ + s;
  unsigned w = 0;
#pragma unroll 8
  for (int j = 0; j < 32; ++j)
    w |= (mrow[(size_t)j * T_] != 0 ? 1u : 0u) << j;
  mkb[(((size_t)b * T_ + s) << 5) + tw] = w;
}

// ---------------------------------------------------------------------------
// GEMM: out[m,n] = sum_c Wb[m,c] * Xt[b,t,c] + bias[m],  n = b*T+t
// Inner loop: two b128 A loads + two b128 B loads + one WMMA per k-chunk.
// mode < 2 : store bf16, scaled, to [b][h][t][32] (q/k/v path)
// mode ==2 : store f32 to [b][m][t] (final output)
// ---------------------------------------------------------------------------
__global__ __launch_bounds__(128) void proj_gemm(
    const __bf16* __restrict__ Wb,   // [192][192] bf16 row-major
    const float*  __restrict__ bias,
    const __bf16* __restrict__ Xt,   // [b][t][192] bf16
    __bf16* __restrict__ outb, float* __restrict__ outf,
    int mode, float oscale)
{
  const int wave = threadIdx.x >> 5;
  const int lane = threadIdx.x & 31;
  const int hf = lane >> 4, ln = lane & 15;
  const int gid = blockIdx.x * 4 + wave;            // 0..3071
  const int mtile = gid % 12, ntile = gid / 12;
  const int m0 = mtile * 16;
  const int n = ntile * 16 + ln;
  const int bb = n >> 10, t = n & (T_ - 1);

  const __bf16* ap = Wb + (m0 + ln) * C_;
  const __bf16* bp = Xt + (size_t)(bb * T_ + t) * C_;

  v8f acc = {};
#pragma unroll
  for (int kc = 0; kc < 6; ++kc) {
    const int c0 = kc * 32 + hf * 8;
    const v16bf a = ld2x16(ap + c0, ap + c0 + 16);
    const v16bf b = ld2x16(bp + c0, bp + c0 + 16);
    acc = __builtin_amdgcn_wmma_f32_16x16x32_bf16(false, a, false, b,
                                                  (short)0, acc, false, false);
  }
#pragma unroll
  for (int vv = 0; vv < 8; ++vv) {
    const int m = m0 + hf * 8 + vv;
    float val = acc[vv] + bias[m];
    if (mode == 2) {
      outf[(bb * C_ + m) * T_ + t] = val;
    } else {
      const int h = m / D_, d = m % D_;
      outb[(((bb * H_ + h) * T_ + t) << 5) + d] = (__bf16)(val * oscale);
    }
  }
}

// ---------------------------------------------------------------------------
// Fused relative attention for one (b,h,256-row segment) per block.
// K rows and V (transposed) staged in LDS; two-pass softmax with WMMA-
// recomputed scores; banded rel-k/rel-v corrections (only 9 offsets matter).
// Mask consumed as packed bits: one b32 load per score tile per lane.
// ---------------------------------------------------------------------------
__global__ __launch_bounds__(256) void attn_kernel(
    const __bf16*   __restrict__ qs,   // [b][h][t][32] bf16, pre-scaled by 1/sqrt(D)
    const __bf16*   __restrict__ ks,   // [b][h][t][32] bf16
    const __bf16*   __restrict__ vs,   // [b][h][t][32] bf16
    const unsigned* __restrict__ mkb,  // [b][s][t/32] row bitmask
    const float*    __restrict__ relk, // [9][24]
    const float*    __restrict__ relv, // [9][24]
    __bf16*         __restrict__ aot)  // [b][t][192] bf16
{
  __shared__ __bf16 ks_s[T_ * KST];     // 80 KB
  __shared__ __bf16 vt_s[DP * VST];     // 65 KB
  __shared__ __bf16 pt_s[8][16 * 32];   // 8 KB  (P tile, per wave)
  __shared__ float  pb_s[8][160];       // band rel-k scores, per wave
  __shared__ float  pe_s[8][160];       // band exp values,  per wave

  const int b   = blockIdx.x >> 5;
  const int h   = (blockIdx.x >> 2) & 7;
  const int seg = blockIdx.x & 3;
  const int tid = threadIdx.x;
  const int wave = tid >> 5, lane = tid & 31;
  const int hf = lane >> 4, ln = lane & 15;

  const __bf16* qsg = qs + (size_t)((b * H_ + h) * T_) * DP;
  const __bf16* ksg = ks + (size_t)((b * H_ + h) * T_) * DP;
  const __bf16* vsg = vs + (size_t)((b * H_ + h) * T_) * DP;
  const unsigned* mkb_b = mkb + ((size_t)b * T_ << 5);

  // stage K ([s][d], padded rows) and V transposed ([d][s]) into LDS
  {
    const unsigned* kg = (const unsigned*)ksg;
    unsigned* kl = (unsigned*)ks_s;
    for (int i = tid; i < T_ * 16; i += 256) {
      const int r = i >> 4, dw = i & 15;
      kl[r * (KST / 2) + dw] = kg[r * 16 + dw];
    }
    for (int i = tid; i < T_ * DP; i += 256) {
      const int d = i & 31, t = i >> 5;
      vt_s[d * VST + t] = vsg[t * DP + d];
    }
  }
  __syncthreads();

  for (int rt = 0; rt < 2; ++rt) {
    const int row0 = seg * 256 + wave * 32 + rt * 16;
    const int wsel = row0 >> 5;              // 32-bit mask word holding all 16 rows
    const int bsh  = (row0 & 31) + hf * 8;   // bit of row (hf*8 + vv=0) in that word

    // precompute banded rel-k scores for the 16 rows (9 offsets each)
    for (int idx = lane; idx < 144; idx += 32) {
      const int r = idx / 9, j = idx % 9;
      const __bf16* qr = qsg + (row0 + r) * DP;
      float s = 0.f;
      for (int d = 0; d < D_; ++d) s += (float)qr[d] * relk[j * D_ + d];
      pb_s[wave][idx] = s;
    }
    for (int idx = lane; idx < 160; idx += 32) pe_s[wave][idx] = 0.f;

    // A operand (Q rows), reused in both passes
    const int am = row0 + ln;
    const v16bf aq = ld2x16(qsg + am * DP + hf * 8, qsg + am * DP + 16 + hf * 8);

    // -------- pass 1: row max --------
    float mx[8];
#pragma unroll
    for (int vv = 0; vv < 8; ++vv) mx[vv] = -3.0e38f;
    for (int ct = 0; ct < 64; ++ct) {
      const __bf16* kb = ks_s + (ct * 16 + ln) * KST;
      const v16bf bk = ld2x16(kb + hf * 8, kb + 16 + hf * 8);
      v8f z = {};
      v8f s8 = __builtin_amdgcn_wmma_f32_16x16x32_bf16(false, aq, false, bk,
                                                       (short)0, z, false, false);
      const int n = ct * 16 + ln;
      const unsigned mw = mkb_b[(n << 5) + wsel];
#pragma unroll
      for (int vv = 0; vv < 8; ++vv) {
        const int r = hf * 8 + vv;
        const int m = row0 + r;
        float s = s8[vv];
        const int dd = n - m;
        if (dd >= -4 && dd <= 4) s += pb_s[wave][r * 9 + dd + 4];
        if (((mw >> (bsh + vv)) & 1u) == 0u) s = -10000.f;
        mx[vv] = fmaxf(mx[vv], s);
      }
    }
#pragma unroll
    for (int off = 1; off < 16; off <<= 1)
#pragma unroll
      for (int vv = 0; vv < 8; ++vv)
        mx[vv] = fmaxf(mx[vv], __shfl_xor(mx[vv], off, 32));

    // -------- pass 2: exp, sum, P·V --------
    float sl[8] = {0.f, 0.f, 0.f, 0.f, 0.f, 0.f, 0.f, 0.f};
    v8f o0 = {}, o1 = {};
    for (int cp = 0; cp < 32; ++cp) {
#pragma unroll
      for (int sub = 0; sub < 2; ++sub) {
        const int ct = cp * 2 + sub;
        const __bf16* kb = ks_s + (ct * 16 + ln) * KST;
        const v16bf bk = ld2x16(kb + hf * 8, kb + 16 + hf * 8);
        v8f z = {};
        v8f s8 = __builtin_amdgcn_wmma_f32_16x16x32_bf16(false, aq, false, bk,
                                                         (short)0, z, false, false);
        const int n = ct * 16 + ln;
        const unsigned mw = mkb_b[(n << 5) + wsel];
#pragma unroll
        for (int vv = 0; vv < 8; ++vv) {
          const int r = hf * 8 + vv;
          const int m = row0 + r;
          float s = s8[vv];
          const int dd = n - m;
          if (dd >= -4 && dd <= 4) s += pb_s[wave][r * 9 + dd + 4];
          if (((mw >> (bsh + vv)) & 1u) == 0u) s = -10000.f;
          const float e = __expf(s - mx[vv]);
          sl[vv] += e;
          if (dd >= -4 && dd <= 4) pe_s[wave][r * 9 + dd + 4] = e;
          pt_s[wave][r * 32 + sub * 16 + ln] = (__bf16)e;
        }
      }
      // P tile (A layout) from wave-private LDS scratch; V from vt_s
      const __bf16* pr = &pt_s[wave][ln * 32];
      const v16bf ap = ld2x16(pr + hf * 8, pr + 16 + hf * 8);
      const __bf16* v0p = vt_s + ln * VST + cp * 32;
      const v16bf bv0 = ld2x16(v0p + hf * 8, v0p + 16 + hf * 8);
      const __bf16* v1p = vt_s + (ln + 16) * VST + cp * 32;
      const v16bf bv1 = ld2x16(v1p + hf * 8, v1p + 16 + hf * 8);
      o0 = __builtin_amdgcn_wmma_f32_16x16x32_bf16(false, ap, false, bv0,
                                                   (short)0, o0, false, false);
      o1 = __builtin_amdgcn_wmma_f32_16x16x32_bf16(false, ap, false, bv1,
                                                   (short)0, o1, false, false);
    }
#pragma unroll
    for (int off = 1; off < 16; off <<= 1)
#pragma unroll
      for (int vv = 0; vv < 8; ++vv)
        sl[vv] += __shfl_xor(sl[vv], off, 32);

    // ---- finalize: normalize + banded rel-v, store bf16 [b][t][192] ----
#pragma unroll
    for (int vv = 0; vv < 8; ++vv) {
      const int r = hf * 8 + vv;
      const int m = row0 + r;
      const float li = 1.0f / sl[vv];
      float a0 = o0[vv] * li;          // d = ln
      float a1 = o1[vv] * li;          // d = ln + 16
#pragma unroll
      for (int j = 0; j < 9; ++j) {
        const float p = pe_s[wave][r * 9 + j] * li;
        a0 += p * relv[j * D_ + ln];
        if (ln < 8) a1 += p * relv[j * D_ + 16 + ln];
      }
      __bf16* orow = aot + (size_t)(b * T_ + m) * C_ + h * D_;
      orow[ln] = (__bf16)a0;
      if (ln < 8) orow[16 + ln] = (__bf16)a1;
    }
  }
}

extern "C" void kernel_launch(void* const* d_in, const int* in_sizes, int n_in,
                              void* d_out, int out_size, void* d_ws, size_t ws_size,
                              hipStream_t stream) {
  (void)in_sizes; (void)n_in; (void)out_size; (void)ws_size;
  const float* x    = (const float*)d_in[0];
  const int*   am   = (const int*)  d_in[1];
  const float* Wq   = (const float*)d_in[2];
  const float* bq   = (const float*)d_in[3];
  const float* Wk   = (const float*)d_in[4];
  const float* bk   = (const float*)d_in[5];
  const float* Wv   = (const float*)d_in[6];
  const float* bv   = (const float*)d_in[7];
  const float* Wo   = (const float*)d_in[8];
  const float* bo   = (const float*)d_in[9];
  const float* relk = (const float*)d_in[10];
  const float* relv = (const float*)d_in[11];

  char* ws = (char*)d_ws;
  __bf16*   qs  = (__bf16*)  (ws);                 // 2 MB   [b][h][t][32]
  __bf16*   ks  = (__bf16*)  (ws + (2u << 20));    // 2 MB
  __bf16*   vs  = (__bf16*)  (ws + (4u << 20));    // 2 MB
  __bf16*   xt  = (__bf16*)  (ws + (6u << 20));    // 1.5 MB [b][t][192]
  __bf16*   aot = (__bf16*)  (ws + (8u << 20));    // 1.5 MB [b][t][192]
  __bf16*   wb  = (__bf16*)  (ws + (10u << 20));   // 0.3 MB 4x[192][192]
  unsigned* mkb = (unsigned*)(ws + (11u << 20));   // 0.5 MB [b][s][t/32]

  // zero q/k/v pad lanes (d=24..31) by zeroing the whole bf16 buffers
  const int nz = (6 << 20) / 4;
  zero_u32<<<(nz + 255) / 256, 256, 0, stream>>>((unsigned*)ws, nz);

  const int nw = 4 * C_ * C_;
  cvt_w<<<(nw + 255) / 256, 256, 0, stream>>>(Wq, Wk, Wv, Wo, wb);
  const int nx = B_ * C_ * T_;
  cvt_x<<<(nx + 255) / 256, 256, 0, stream>>>(x, xt);
  const int nm = B_ * T_ * 32;
  pack_mask<<<(nm + 255) / 256, 256, 0, stream>>>(am, mkb);

  const float qscale = 0.20412414523193154f;   // 1/sqrt(24)
  proj_gemm<<<768, 128, 0, stream>>>(wb + 0 * C_ * C_, bq, xt, qs, nullptr, 0, qscale);
  proj_gemm<<<768, 128, 0, stream>>>(wb + 1 * C_ * C_, bk, xt, ks, nullptr, 0, 1.0f);
  proj_gemm<<<768, 128, 0, stream>>>(wb + 2 * C_ * C_, bv, xt, vs, nullptr, 0, 1.0f);

  attn_kernel<<<B_ * H_ * 4, 256, 0, stream>>>(qs, ks, vs, mkb, relk, relv, aot);

  proj_gemm<<<768, 128, 0, stream>>>(wb + 3 * C_ * C_, bo, aot, nullptr,
                                     (float*)d_out, 2, 1.0f);
}
